// ZoneD_77163382440508
// MI455X (gfx1250) — matrix operand order repaired and checked by reading, not compiled, not run on hardware
//
#include <hip/hip_runtime.h>
#include <hip/hip_bf16.h>

#define DEVI __device__ __forceinline__

typedef __bf16 bf16_t;
typedef __bf16 v16bf __attribute__((ext_vector_type(16)));
typedef __bf16 v8bf  __attribute__((ext_vector_type(8)));
typedef float  v8f   __attribute__((ext_vector_type(8)));

// Problem constants (match setup_inputs)
constexpr int Bb = 8, Ll = 2048, Mm = 256, Dd = 1024, DO = 256;
constexpr int BL = Bb * Ll;                 // 16384 rows for the big GEMMs
constexpr int NCH = 8, CHUNK = Ll / NCH;    // chunked scan

DEVI float sigmoidf(float x) { return 1.0f / (1.0f + __expf(-x)); }
DEVI bf16_t tobf(float f)    { return (bf16_t)f; }

// ---------------------------------------------------------------------------
// fp32 -> bf16 cast
// ---------------------------------------------------------------------------
__global__ __launch_bounds__(256) void k_cast_bf16(const float* __restrict__ x,
                                                   bf16_t* __restrict__ y, int n) {
  int i = blockIdx.x * 256 + threadIdx.x;
  if (i < n) y[i] = tobf(x[i]);
}

// ---------------------------------------------------------------------------
// EMA smoothing over boundaries: sequential over M, parallel over (B, d)
// ---------------------------------------------------------------------------
__global__ __launch_bounds__(256) void k_ema(const float* __restrict__ concept_out,
                                             const float* __restrict__ probs,
                                             const int* __restrict__ bidx,
                                             float* __restrict__ smoothed) {
  int idx = blockIdx.x * 256 + threadIdx.x;   // over B*Dd
  int b = idx / Dd, c = idx % Dd;
  float h = 0.f;
  for (int m = 0; m < Mm; ++m) {
    int bi = bidx[b * Mm + m];
    float p = probs[b * Ll + bi];
    float ci = concept_out[((size_t)b * Mm + m) * Dd + c];
    h = p * ci + (1.f - p) * h;
    smoothed[((size_t)b * Mm + m) * Dd + c] = h;
  }
}

// ---------------------------------------------------------------------------
// bucket[b,l] = clip(searchsorted_right(boundary_idx[b], l) - 1, 0)
// ---------------------------------------------------------------------------
__global__ __launch_bounds__(256) void k_bucket(const int* __restrict__ bidx,
                                                int* __restrict__ bucket) {
  int idx = blockIdx.x * 256 + threadIdx.x;   // over B*Ll
  if (idx >= Bb * Ll) return;
  int b = idx / Ll, l = idx % Ll;
  const int* row = bidx + b * Mm;
  int lo = 0, hi = Mm;
  while (lo < hi) { int mid = (lo + hi) >> 1; if (row[mid] <= l) lo = mid + 1; else hi = mid; }
  int bk = lo - 1; if (bk < 0) bk = 0;
  bucket[idx] = bk;
}

// ---------------------------------------------------------------------------
// WMMA bf16 GEMM: D[M,N] = epi(A[M,K] @ W[N,K]^T + bias)
// block = 256 threads (8 waves). Wave tile = 32x64 (2 row strips x 4 col
// tiles, 8 WMMAs/k-step reusing 4 B fragments). Block tile = 256x64.
// B tile (64x32 bf16 = 4KB) staged in LDS via global_load_async_to_lds_b128,
// double buffered, synced with s_wait_asynccnt + workgroup barrier.
// EPI: 0 = identity, 1 = sigmoid
// ---------------------------------------------------------------------------
union AFrag { v16bf v; v8bf h[2]; };
union CFrag { v8f v; float e[8]; };

template <int EPI>
__global__ __launch_bounds__(256) void k_gemm(const bf16_t* __restrict__ A,
                                              const bf16_t* __restrict__ W,
                                              const float* __restrict__ bias,
                                              float* __restrict__ D,
                                              int M, int N, int K) {
  constexpr int ROWP = 40;  // bf16 elems per LDS row: 80B pitch (16B-aligned, 20-bank stride)
  __shared__ __align__(16) bf16_t smem[2][64 * ROWP];

  const int tid  = threadIdx.x;
  const int lane = tid & 31;
  const int wave = tid >> 5;
  const int hf   = lane >> 4;       // 0 or 1 (K-half select)
  const int l16  = lane & 15;       // row (A) / col (B) within tile
  const int row0 = blockIdx.x * 256 + wave * 32;
  const int col0 = blockIdx.y * 64;

  // cooperative B-tile staging: thread -> (tile row bn, 16B quarter bq)
  const int bn = tid >> 2;          // 0..63  (output column within tile)
  const int bq = tid & 3;           // 0..3   (8 bf16 = 16B each)
  const bf16_t* gB = W + (size_t)(col0 + bn) * K + bq * 8;

  auto stageB = [&](int buf, int kk) {
    // low 32 bits of a generic LDS address == LDS byte offset (ISA aperture rule)
    unsigned dst = (unsigned)(uintptr_t)(&smem[buf][bn * ROWP + bq * 8]);
    const bf16_t* src = gB + kk;
    asm volatile("global_load_async_to_lds_b128 %0, %1, off"
                 :: "v"(dst), "v"(src) : "memory");
  };

  CFrag acc0[4] = {}, acc1[4] = {};
  const bf16_t* arow0 = A + (size_t)(row0 + l16) * K;
  const bf16_t* arow1 = arow0 + (size_t)16 * K;

  stageB(0, 0);
  int cur = 0;
  for (int kk = 0; kk < K; kk += 32) {
#if __has_builtin(__builtin_amdgcn_s_wait_asynccnt)
    __builtin_amdgcn_s_wait_asynccnt(0);
#else
    asm volatile("s_wait_asynccnt 0x0" ::: "memory");
#endif
    __syncthreads();
    if (kk + 32 < K) stageB(cur ^ 1, kk + 32);

    AFrag a0, a1;
    a0.h[0] = *(const v8bf*)(arow0 + kk + 16 * hf);
    a0.h[1] = *(const v8bf*)(arow0 + kk + 16 * hf + 8);
    a1.h[0] = *(const v8bf*)(arow1 + kk + 16 * hf);
    a1.h[1] = *(const v8bf*)(arow1 + kk + 16 * hf + 8);
    if (kk + 128 < K) {
      __builtin_prefetch(arow0 + kk + 128, 0, 1);  // global_prefetch_b8
      __builtin_prefetch(arow1 + kk + 128, 0, 1);
    }
    // hoist all B fragments so one dscnt wait covers the whole WMMA burst
    v16bf bfr[4];
#pragma unroll
    for (int t = 0; t < 4; ++t)
      bfr[t] = *(const v16bf*)(&smem[cur][(t * 16 + l16) * ROWP + 16 * hf]);
#pragma unroll
    for (int t = 0; t < 4; ++t)
      acc0[t].v = __builtin_amdgcn_wmma_f32_16x16x32_bf16(
          false, a0.v, false, bfr[t], (short)0, acc0[t].v, false, false);
#pragma unroll
    for (int t = 0; t < 4; ++t)
      acc1[t].v = __builtin_amdgcn_wmma_f32_16x16x32_bf16(
          false, a1.v, false, bfr[t], (short)0, acc1[t].v, false, false);
    cur ^= 1;
  }

#pragma unroll
  for (int t = 0; t < 4; ++t) {
    int col = col0 + t * 16 + l16;
    float bv = bias ? bias[col] : 0.f;
#pragma unroll
    for (int r = 0; r < 8; ++r) {
      int row = row0 + r + 8 * hf;
      float v0 = acc0[t].e[r] + bv;
      float v1 = acc1[t].e[r] + bv;
      if (EPI == 1) { v0 = sigmoidf(v0); v1 = sigmoidf(v1); }
      D[(size_t)row * N + col] = v0;
      D[(size_t)(row + 16) * N + col] = v1;
    }
  }
}

// ---------------------------------------------------------------------------
// gated residual + plug-back gather:
// h = (1-p)*sigmoid(gate)*enc + smoothed[b, bucket[b,l], :]
// writes fp32 (in place over gsig) and bf16 copy for the down GEMM
// ---------------------------------------------------------------------------
__global__ __launch_bounds__(256) void k_fuse_gate(const float* __restrict__ gsig,
                                                   const float* __restrict__ enc,
                                                   const float* __restrict__ probs,
                                                   const int* __restrict__ bucket,
                                                   const float* __restrict__ smoothed,
                                                   float* __restrict__ hf,
                                                   bf16_t* __restrict__ hbf) {
  size_t idx = (size_t)blockIdx.x * 256 + threadIdx.x;   // over B*Ll*Dd
  int c  = (int)(idx % Dd);
  int bl = (int)(idx / Dd);
  int b  = bl / Ll;
  float p  = probs[bl];
  float pb = smoothed[((size_t)b * Mm + bucket[bl]) * Dd + c];
  float v  = (1.f - p) * gsig[idx] * enc[idx] + pb;
  hf[idx]  = v;
  hbf[idx] = tobf(v);
}

// ---------------------------------------------------------------------------
// depthwise causal conv (k=4) + bias; emits fp32 and bf16
// ---------------------------------------------------------------------------
__global__ __launch_bounds__(256) void k_conv(const float* __restrict__ x,
                                              const float* __restrict__ cw,  // [DO,4]
                                              const float* __restrict__ cb,  // [DO]
                                              float* __restrict__ yf,
                                              bf16_t* __restrict__ ybf) {
  size_t idx = (size_t)blockIdx.x * 256 + threadIdx.x;   // over B*Ll*DO
  int c  = (int)(idx % DO);
  int bl = (int)(idx / DO);
  int l  = bl % Ll;
  float acc = cb[c];
#pragma unroll
  for (int t = 0; t < 4; ++t) {
    int ll = l - 3 + t;
    if (ll >= 0) acc += cw[c * 4 + t] * x[idx + (size_t)(ll - l) * DO];
  }
  yf[idx]  = acc;
  ybf[idx] = tobf(acc);
}

// ---------------------------------------------------------------------------
// a_t = sigmoid(log_a)^(8*r) ; u_t = i * x_conv   (in-place over r/i buffers)
// ---------------------------------------------------------------------------
__global__ __launch_bounds__(256) void k_gates(const float* __restrict__ log_a,
                                               const float* __restrict__ xc,
                                               float* __restrict__ r_a,   // in: sigmoid(r), out: a_t
                                               float* __restrict__ i_u) { // in: sigmoid(i), out: u_t
  size_t idx = (size_t)blockIdx.x * 256 + threadIdx.x;   // over B*Ll*DO
  int c = (int)(idx % DO);
  float l2 = __log2f(sigmoidf(log_a[c]));
  float a  = exp2f(8.f * r_a[idx] * l2);
  r_a[idx] = a;
  i_u[idx] = i_u[idx] * xc[idx];
}

// ---------------------------------------------------------------------------
// chunked linear-recurrence scan: h = a*h + sqrt(max(1-a*a,0))*u
// ---------------------------------------------------------------------------
__global__ __launch_bounds__(256) void k_scan_p1(const float* __restrict__ a_t,
                                                 const float* __restrict__ u_t,
                                                 float* __restrict__ carryA,
                                                 float* __restrict__ carryH) {
  int idx = blockIdx.x * 256 + threadIdx.x;  // over B*NCH*DO
  int c = idx % DO, j = (idx / DO) % NCH, b = idx / (DO * NCH);
  const float* ap = a_t + ((size_t)b * Ll + j * CHUNK) * DO + c;
  const float* up = u_t + ((size_t)b * Ll + j * CHUNK) * DO + c;
  float A = 1.f, h = 0.f;
#pragma unroll 4
  for (int i = 0; i < CHUNK; ++i) {
    float a = ap[(size_t)i * DO];
    float u = up[(size_t)i * DO];
    float g = __fsqrt_rn(fmaxf(1.f - a * a, 0.f));
    h = a * h + g * u;
    A *= a;
  }
  carryA[(b * NCH + j) * DO + c] = A;
  carryH[(b * NCH + j) * DO + c] = h;
}

__global__ __launch_bounds__(256) void k_scan_p2(const float* __restrict__ carryA,
                                                 const float* __restrict__ carryH,
                                                 float* __restrict__ hstart) {
  int idx = blockIdx.x * 256 + threadIdx.x;  // over B*DO
  int c = idx % DO, b = idx / DO;
  float h = 0.f;
  for (int j = 0; j < NCH; ++j) {
    hstart[(b * NCH + j) * DO + c] = h;
    h = carryA[(b * NCH + j) * DO + c] * h + carryH[(b * NCH + j) * DO + c];
  }
}

__global__ __launch_bounds__(256) void k_scan_p3(const float* __restrict__ a_t,
                                                 const float* __restrict__ u_t,
                                                 const float* __restrict__ hstart,
                                                 bf16_t* __restrict__ sbf) {
  int idx = blockIdx.x * 256 + threadIdx.x;  // over B*NCH*DO
  int c = idx % DO, j = (idx / DO) % NCH, b = idx / (DO * NCH);
  const float* ap = a_t + ((size_t)b * Ll + j * CHUNK) * DO + c;
  const float* up = u_t + ((size_t)b * Ll + j * CHUNK) * DO + c;
  bf16_t* sp = sbf + ((size_t)b * Ll + j * CHUNK) * DO + c;
  float h = hstart[(b * NCH + j) * DO + c];
#pragma unroll 4
  for (int i = 0; i < CHUNK; ++i) {
    float a = ap[(size_t)i * DO];
    float u = up[(size_t)i * DO];
    float g = __fsqrt_rn(fmaxf(1.f - a * a, 0.f));
    h = a * h + g * u;
    sp[(size_t)i * DO] = tobf(h);
  }
}

// ---------------------------------------------------------------------------
// RMSNorm over last dim n (256 or 1024); one block per row; wave32 reduction
// ---------------------------------------------------------------------------
__global__ __launch_bounds__(256) void k_rmsnorm(const float* __restrict__ x,
                                                 const float* __restrict__ w,
                                                 float* __restrict__ yf,
                                                 bf16_t* __restrict__ ybf, int n) {
  __shared__ float red[8];
  const int row = blockIdx.x;
  const int tid = threadIdx.x;
  const int per = n >> 8;                 // 1 (n=256) or 4 (n=1024)
  const float* xr = x + (size_t)row * n;
  float xv[4];
  float ss = 0.f;
  for (int k = 0; k < per; ++k) { float v = xr[tid + k * 256]; xv[k] = v; ss += v * v; }
#pragma unroll
  for (int off = 16; off > 0; off >>= 1) ss += __shfl_xor(ss, off, 32);
  if ((tid & 31) == 0) red[tid >> 5] = ss;
  __syncthreads();
  if (tid == 0) {
    float t = 0.f;
    for (int i = 0; i < 8; ++i) t += red[i];
    red[0] = rsqrtf(t / (float)n + 1e-6f);
  }
  __syncthreads();
  float sc = red[0];
  for (int k = 0; k < per; ++k) {
    int col = tid + k * 256;
    float v = xv[k] * sc * w[col];
    yf[(size_t)row * n + col] = v;
    if (ybf) ybf[(size_t)row * n + col] = tobf(v);
  }
}

// ---------------------------------------------------------------------------
// host orchestration
// ---------------------------------------------------------------------------
static inline void* carve(char*& p, size_t bytes) {
  void* r = (void*)p;
  p += (bytes + 255) & ~(size_t)255;
  return r;
}

extern "C" void kernel_launch(void* const* d_in, const int* in_sizes, int n_in,
                              void* d_out, int out_size, void* d_ws, size_t ws_size,
                              hipStream_t stream) {
  const float* concept_out = (const float*)d_in[0];
  const float* encoder_out = (const float*)d_in[1];
  const float* bprobs      = (const float*)d_in[2];
  const int*   bidx        = (const int*)d_in[3];
  const float* gate_w      = (const float*)d_in[4];
  const float* gate_b      = (const float*)d_in[5];
  const float* down_w      = (const float*)d_in[6];
  const float* up_w        = (const float*)d_in[7];
  const float* norm_out_w  = (const float*)d_in[8];
  const float* rec_conv_w  = (const float*)d_in[9];
  const float* rec_conv_b  = (const float*)d_in[10];
  const float* rec_Wr_w    = (const float*)d_in[11];
  const float* rec_Wr_b    = (const float*)d_in[12];
  const float* rec_Wi_w    = (const float*)d_in[13];
  const float* rec_Wi_b    = (const float*)d_in[14];
  const float* rec_log_a   = (const float*)d_in[15];
  const float* rec_out_w   = (const float*)d_in[16];
  const float* rec_norm_w  = (const float*)d_in[17];
  float* out = (float*)d_out;

  char* p = (char*)d_ws;
  float*  smoothed = (float*) carve(p, (size_t)Bb * Mm * Dd * 4);
  int*    bucket   = (int*)   carve(p, (size_t)Bb * Ll * 4);
  bf16_t* enc_bf   = (bf16_t*)carve(p, (size_t)BL * Dd * 2);
  bf16_t* wg_bf    = (bf16_t*)carve(p, (size_t)Dd * Dd * 2);
  bf16_t* wd_bf    = (bf16_t*)carve(p, (size_t)DO * Dd * 2);
  bf16_t* wu_bf    = (bf16_t*)carve(p, (size_t)Dd * DO * 2);
  bf16_t* wr_bf    = (bf16_t*)carve(p, (size_t)3 * DO * DO * 2);
  bf16_t* wi_bf    = (bf16_t*)carve(p, (size_t)3 * DO * DO * 2);
  bf16_t* wo_bf    = (bf16_t*)carve(p, (size_t)3 * DO * DO * 2);
  float*  gsig     = (float*) carve(p, (size_t)BL * Dd * 4);   // reused: h_big, up_out
  bf16_t* hbig_bf  = (bf16_t*)carve(p, (size_t)BL * Dd * 2);
  float*  xc_f     = (float*) carve(p, (size_t)BL * DO * 4);
  bf16_t* xc_bf    = (bf16_t*)carve(p, (size_t)BL * DO * 2);
  float*  r_a      = (float*) carve(p, (size_t)BL * DO * 4);   // sigmoid(r) -> a_t
  float*  i_u      = (float*) carve(p, (size_t)BL * DO * 4);   // sigmoid(i) -> u_t
  bf16_t* s_bf     = (bf16_t*)carve(p, (size_t)BL * DO * 2);
  float*  o_f      = (float*) carve(p, (size_t)BL * DO * 4);
  float*  h_f      = (float*) carve(p, (size_t)BL * DO * 4);
  bf16_t* h_bf     = (bf16_t*)carve(p, (size_t)BL * DO * 2);
  float*  carryA   = (float*) carve(p, (size_t)Bb * NCH * DO * 4);
  float*  carryH   = (float*) carve(p, (size_t)Bb * NCH * DO * 4);
  float*  hstart   = (float*) carve(p, (size_t)Bb * NCH * DO * 4);

  auto cast = [&](const float* src, bf16_t* dst, size_t n) {
    k_cast_bf16<<<(unsigned)((n + 255) / 256), 256, 0, stream>>>(src, dst, (int)n);
  };

  // 0) bf16 copies of activations + weights
  cast(encoder_out, enc_bf, (size_t)BL * Dd);
  cast(gate_w, wg_bf, (size_t)Dd * Dd);
  cast(down_w, wd_bf, (size_t)DO * Dd);
  cast(up_w,   wu_bf, (size_t)Dd * DO);
  cast(rec_Wr_w,  wr_bf, (size_t)3 * DO * DO);
  cast(rec_Wi_w,  wi_bf, (size_t)3 * DO * DO);
  cast(rec_out_w, wo_bf, (size_t)3 * DO * DO);

  // 1) EMA over boundaries ; 2) plug-back buckets
  k_ema<<<(Bb * Dd) / 256, 256, 0, stream>>>(concept_out, bprobs, bidx, smoothed);
  k_bucket<<<(Bb * Ll + 255) / 256, 256, 0, stream>>>(bidx, bucket);

  // 3) gate GEMM (sigmoid epilogue) + gated residual fuse
  k_gemm<1><<<dim3(BL / 256, Dd / 64), 256, 0, stream>>>(enc_bf, wg_bf, gate_b, gsig, BL, Dd, Dd);
  k_fuse_gate<<<(unsigned)(((size_t)BL * Dd) / 256), 256, 0, stream>>>(
      gsig, encoder_out, bprobs, bucket, smoothed, gsig, hbig_bf);

  // 4) down GEMM -> h_f [BL, DO]
  k_gemm<0><<<dim3(BL / 256, DO / 64), 256, 0, stream>>>(hbig_bf, wd_bf, nullptr, h_f, BL, DO, Dd);

  // 5) three recurrence layers
  for (int k = 0; k < 3; ++k) {
    k_conv<<<(unsigned)(((size_t)BL * DO) / 256), 256, 0, stream>>>(
        h_f, rec_conv_w + (size_t)k * DO * 4, rec_conv_b + (size_t)k * DO, xc_f, xc_bf);
    k_gemm<1><<<dim3(BL / 256, DO / 64), 256, 0, stream>>>(
        xc_bf, wr_bf + (size_t)k * DO * DO, rec_Wr_b + (size_t)k * DO, r_a, BL, DO, DO);
    k_gemm<1><<<dim3(BL / 256, DO / 64), 256, 0, stream>>>(
        xc_bf, wi_bf + (size_t)k * DO * DO, rec_Wi_b + (size_t)k * DO, i_u, BL, DO, DO);
    k_gates<<<(unsigned)(((size_t)BL * DO) / 256), 256, 0, stream>>>(
        rec_log_a + (size_t)k * DO, xc_f, r_a, i_u);
    k_scan_p1<<<(Bb * NCH * DO) / 256, 256, 0, stream>>>(r_a, i_u, carryA, carryH);
    k_scan_p2<<<(Bb * DO) / 256, 256, 0, stream>>>(carryA, carryH, hstart);
    k_scan_p3<<<(Bb * NCH * DO) / 256, 256, 0, stream>>>(r_a, i_u, hstart, s_bf);
    k_gemm<0><<<dim3(BL / 256, DO / 64), 256, 0, stream>>>(
        s_bf, wo_bf + (size_t)k * DO * DO, nullptr, o_f, BL, DO, DO);
    k_rmsnorm<<<BL, 256, 0, stream>>>(o_f, rec_norm_w + (size_t)k * DO, h_f, h_bf, DO);
  }

  // 6) up GEMM -> [BL, Dd], final RMSNorm -> d_out
  k_gemm<0><<<dim3(BL / 256, Dd / 64), 256, 0, stream>>>(h_bf, wu_bf, nullptr, gsig, BL, Dd, DO);
  k_rmsnorm<<<BL, 256, 0, stream>>>(gsig, norm_out_w, out, nullptr, Dd);
}
